// RSSM_15556371546796
// MI455X (gfx1250) — matrix operand order repaired and checked by reading, not compiled, not run on hardware
//
#include <hip/hip_runtime.h>
#include <cmath>

typedef __attribute__((ext_vector_type(16))) __bf16 v16bf;
typedef __attribute__((ext_vector_type(8)))  __bf16 v8bf;
typedef __attribute__((ext_vector_type(8)))  float  v8f;

#define TT 50
#define BB 8192
#define ADIM 6
#define SDIM 200
#define LDIM 30
#define KP 224      // padded K for 200
#define CK 64       // padded K for concat(30+6)
#define HP 208      // padded N for 200
#define BM 64       // batch rows per workgroup (4 M-tiles)
#define NTHREADS 256  // 8 waves; 2 round-groups of 4 waves; each wave owns 2 M-tiles

#define WMMA_BF16(A, Bf, C) \
  __builtin_amdgcn_wmma_f32_16x16x32_bf16(false, (A), false, (Bf), (short)0, (C), false, false)

// ---- fragment loaders -------------------------------------------------------
// A (16x32 bf16, MxK): lane m = lane&15, g = lane>>4.
//   halves 0..15 -> K = ks*32 + 16g + 0..15
__device__ __forceinline__ v16bf ldA(const __bf16* base, int stride, int mt, int lane, int ks) {
  int m = mt * 16 + (lane & 15);
  int g = lane >> 4;
  const __bf16* p = base + m * stride + ks * 32 + g * 16;
  v8bf lo = *(const v8bf*)(p);
  v8bf hi = *(const v8bf*)(p + 8);
  return __builtin_shufflevector(lo, hi, 0,1,2,3,4,5,6,7,8,9,10,11,12,13,14,15);
}

// B (32x16 bf16, KxN) sourced from row-major W[n][k] (computes X @ W^T):
//   lane n = lane&15, g = lane>>4; halves 0..15 -> K = ks*32 + 16g + 0..15
__device__ __forceinline__ v16bf ldB(const __bf16* W, int stride, int nt, int lane, int ks) {
  int n = nt * 16 + (lane & 15);
  int g = lane >> 4;
  return *(const v16bf*)(W + n * stride + ks * 32 + g * 16);
}

// ---- weight / bias padding (once per launch) -------------------------------
__global__ void pad_w(const float* __restrict__ src, __bf16* __restrict__ dst,
                      int rows, int cols, int prows, int pcols) {
  int idx = blockIdx.x * blockDim.x + threadIdx.x;
  if (idx >= prows * pcols) return;
  int r = idx / pcols, c = idx % pcols;
  float v = (r < rows && c < cols) ? src[r * cols + c] : 0.0f;
  dst[idx] = (__bf16)v;
}

__global__ void pad_bias(const float* __restrict__ src, float* __restrict__ dst, int n, int pn) {
  int i = blockIdx.x * blockDim.x + threadIdx.x;
  if (i < pn) dst[i] = (i < n) ? src[i] : 0.0f;
}

// ---- fused RSSM step --------------------------------------------------------
__global__ __launch_bounds__(NTHREADS)
void rssm_step(int t,
               const float* __restrict__ actions, const float* __restrict__ h0,
               const float* __restrict__ s0, const float* __restrict__ noise,
               const __bf16* __restrict__ Wfc, const float* __restrict__ bfc,
               const __bf16* __restrict__ Wih, const float* __restrict__ bih,
               const __bf16* __restrict__ Whh, const float* __restrict__ bhh,
               const __bf16* __restrict__ W1w, const float* __restrict__ b1,
               const __bf16* __restrict__ Wmu, const float* __restrict__ bmu,
               const __bf16* __restrict__ Wvr, const float* __restrict__ bvr,
               float* __restrict__ states, float* __restrict__ stoch) {
  __shared__ __align__(32) __bf16 A1[BM * CK];       //  8 KB  concat(s,a)
  __shared__ __align__(32) __bf16 Xbf[BM * KP];      // 28 KB  fc output / later: hid
  __shared__ __align__(32) __bf16 Hbf[BM * KP];      // 28 KB  h_prev
  __shared__ __align__(32) __bf16 HN[BM * KP];       // 28 KB  h_new
  __shared__ __align__(32) float  Gb[2 * BM * 64];   // 32 KB  gate buffers (per group) / mu-var

  const int tid  = threadIdx.x;
  const int lane = tid & 31;
  const int wv   = tid >> 5;
  const int row0 = blockIdx.x * BM;
  const int n16  = lane & 15;
  const int g    = lane >> 4;

  const float* hprev = (t == 0) ? (h0 + (size_t)row0 * SDIM)
                                : (states + (size_t)(t - 1) * BB * SDIM + (size_t)row0 * SDIM);
  const float* sprev = (t == 0) ? (s0 + (size_t)row0 * LDIM)
                                : (stoch + (size_t)(t - 1) * BB * LDIM + (size_t)row0 * LDIM);
  const float* act = actions + (size_t)t * BB * ADIM + (size_t)row0 * ADIM;
  const float* eps = noise + (size_t)t * BB * LDIM + (size_t)row0 * LDIM;
  float* hout = states + (size_t)t * BB * SDIM + (size_t)row0 * SDIM;
  float* sout = stoch + (size_t)t * BB * LDIM + (size_t)row0 * LDIM;

  // ---- stage activations into LDS (bf16), zero pad tails -------------------
  for (int i = tid; i < BM * CK; i += NTHREADS) {
    int m = i >> 6, k = i & 63;
    float v = 0.f;
    if (k < LDIM) v = sprev[m * LDIM + k];
    else if (k < LDIM + ADIM) v = act[m * ADIM + (k - LDIM)];
    A1[i] = (__bf16)v;
  }
  for (int i = tid; i < BM * KP; i += NTHREADS) {
    int m = i / KP, k = i % KP;
    float v = (k < SDIM) ? hprev[m * SDIM + k] : 0.f;
    Hbf[i] = (__bf16)v;
  }
  for (int i = tid; i < BM * (KP - SDIM); i += NTHREADS) {  // zero K-pad cols
    int m = i / (KP - SDIM), k = SDIM + i % (KP - SDIM);
    Xbf[m * KP + k] = (__bf16)0.f;
    HN[m * KP + k]  = (__bf16)0.f;
  }
  __syncthreads();

  const int mtb  = (wv & 1) * 2;       // this wave owns M-tiles mtb, mtb+1
  const int ctg  = wv >> 1;            // phase A/C column-tile group (0..3)
  const int grp  = wv >> 2;            // phase-B round group (0: even jr, 1: odd jr)
  const int pair = (wv >> 1) & 1;      // phase-B: 0 -> (r, gi_n), 1 -> (z, gh_n)
  const int tid2 = tid & 127;          // thread id within round group

  // ---- Phase A: X = concat(s,a) @ Wfc^T + bfc  (biases pre-padded) ---------
  {
    v16bf a1f[2][2];
#pragma unroll
    for (int mi = 0; mi < 2; ++mi)
#pragma unroll
      for (int ks = 0; ks < 2; ++ks) a1f[mi][ks] = ldA(A1, CK, mtb + mi, lane, ks);
    for (int ct = ctg; ct < 13; ct += 4) {
      v8f acc[2] = {{}, {}};
#pragma unroll
      for (int ks = 0; ks < 2; ++ks) {
        v16bf b = ldB(Wfc, CK, ct, lane, ks);
#pragma unroll
        for (int mi = 0; mi < 2; ++mi) acc[mi] = WMMA_BF16(a1f[mi][ks], b, acc[mi]);
      }
      int n = ct * 16 + n16;
      float bias = bfc[n];  // padded: zero for n >= 200
#pragma unroll
      for (int mi = 0; mi < 2; ++mi)
#pragma unroll
        for (int j = 0; j < 8; ++j)
          Xbf[((mtb + mi) * 16 + g * 8 + j) * KP + n] = (__bf16)(acc[mi][j] + bias);
    }
  }
  __syncthreads();

  // ---- Phase B: GRU gates; group 0 takes even rounds, group 1 odd ----------
  {
    v16bf axf[2][7], ahf[2][7];
#pragma unroll
    for (int mi = 0; mi < 2; ++mi)
#pragma unroll
      for (int ks = 0; ks < 7; ++ks) {
        axf[mi][ks] = ldA(Xbf, KP, mtb + mi, lane, ks);
        ahf[mi][ks] = ldA(Hbf, KP, mtb + mi, lane, ks);
      }
    float* Gg = Gb + grp * BM * 64;
    for (int it = 0; it < 7; ++it) {
      int jr = it * 2 + grp;      // group 1, it==6 -> jr==13: dummy round (stores guarded)
      v8f accRZ[2] = {{}, {}};    // r (pair 0) or z (pair 1): X@Wih + H@Whh merged
      v8f accN[2]  = {{}, {}};    // gi_n (pair 0) or gh_n (pair 1)
      int ntRZ = pair * 13 + jr;  // r rows 0..207, z rows 208..415
      int ntN  = 26 + jr;         // n rows 416..623
      if (pair == 0) {            // wave-uniform branch
#pragma unroll
        for (int ks = 0; ks < 7; ++ks) {
          v16bf b0 = ldB(Wih, KP, ntRZ, lane, ks);
          v16bf b1f = ldB(Whh, KP, ntRZ, lane, ks);
          v16bf b2 = ldB(Wih, KP, ntN, lane, ks);
#pragma unroll
          for (int mi = 0; mi < 2; ++mi) {
            accRZ[mi] = WMMA_BF16(axf[mi][ks], b0, accRZ[mi]);
            accRZ[mi] = WMMA_BF16(ahf[mi][ks], b1f, accRZ[mi]);
            accN[mi]  = WMMA_BF16(axf[mi][ks], b2, accN[mi]);
          }
        }
      } else {
#pragma unroll
        for (int ks = 0; ks < 7; ++ks) {
          v16bf b0 = ldB(Wih, KP, ntRZ, lane, ks);
          v16bf b1f = ldB(Whh, KP, ntRZ, lane, ks);
          v16bf b2 = ldB(Whh, KP, ntN, lane, ks);
#pragma unroll
          for (int mi = 0; mi < 2; ++mi) {
            accRZ[mi] = WMMA_BF16(axf[mi][ks], b0, accRZ[mi]);
            accRZ[mi] = WMMA_BF16(ahf[mi][ks], b1f, accRZ[mi]);
            accN[mi]  = WMMA_BF16(ahf[mi][ks], b2, accN[mi]);
          }
        }
      }
#pragma unroll
      for (int mi = 0; mi < 2; ++mi)
#pragma unroll
        for (int j = 0; j < 8; ++j) {
          int m = (mtb + mi) * 16 + g * 8 + j;
          Gg[m * 64 + pair * 16 + n16]      = accRZ[mi][j];  // 0..15 r, 16..31 z
          Gg[m * 64 + 32 + pair * 16 + n16] = accN[mi][j];   // 32..47 gi_n, 48..63 gh_n
        }
      __syncthreads();
      for (int e = tid2; e < BM * 16; e += 128) {
        int m = e >> 4, c = e & 15;
        int col = jr * 16 + c;  // 0..223; biases padded (dummy round reads stay in d_ws)
        float gr = Gg[m * 64 + c]      + bih[col] + bhh[col];
        float gz = Gg[m * 64 + 16 + c] + bih[HP + col] + bhh[HP + col];
        float r = 1.f / (1.f + __expf(-gr));
        float z = 1.f / (1.f + __expf(-gz));
        float gn = (Gg[m * 64 + 32 + c] + bih[2 * HP + col]) +
                   r * (Gg[m * 64 + 48 + c] + bhh[2 * HP + col]);
        float nn = tanhf(gn);
        bool ok = (col < SDIM);
        float hp = ok ? hprev[m * SDIM + col] : 0.f;
        float hn = (1.f - z) * nn + z * hp;
        if (ok) {
          hout[m * SDIM + col] = hn;
          HN[m * KP + col] = (__bf16)hn;
        }
      }
      __syncthreads();
    }
  }

  // ---- Phase C: hid = relu(h_new @ W1^T + b1) -> Xbf (reused) --------------
  {
    v16bf hnf[2][7];
#pragma unroll
    for (int mi = 0; mi < 2; ++mi)
#pragma unroll
      for (int ks = 0; ks < 7; ++ks) hnf[mi][ks] = ldA(HN, KP, mtb + mi, lane, ks);
    for (int ct = ctg; ct < 13; ct += 4) {
      v8f acc[2] = {{}, {}};
#pragma unroll
      for (int ks = 0; ks < 7; ++ks) {
        v16bf b = ldB(W1w, KP, ct, lane, ks);
#pragma unroll
        for (int mi = 0; mi < 2; ++mi) acc[mi] = WMMA_BF16(hnf[mi][ks], b, acc[mi]);
      }
      int n = ct * 16 + n16;
      float bias = b1[n];  // padded: zero for n >= 200
#pragma unroll
      for (int mi = 0; mi < 2; ++mi)
#pragma unroll
        for (int j = 0; j < 8; ++j) {
          float v = acc[mi][j] + bias;
          v = (v > 0.f) ? v : 0.f;  // relu(0+0)=0 keeps pad cols zero
          Xbf[((mtb + mi) * 16 + g * 8 + j) * KP + n] = (__bf16)v;
        }
    }
  }
  __syncthreads();

  // ---- Phase D: mu / var heads (one head-tile per wave), then s_new --------
  {
    v16bf xf[2][7];
#pragma unroll
    for (int mi = 0; mi < 2; ++mi)
#pragma unroll
      for (int ks = 0; ks < 7; ++ks) xf[mi][ks] = ldA(Xbf, KP, mtb + mi, lane, ks);
    int job = wv >> 1;                      // 0..3
    int ct = job & 1;                       // latent col-tile (LATENT padded to 32)
    const __bf16* Wh = (job >> 1) ? Wvr : Wmu;
    int colbase = (job >> 1) * 32 + ct * 16;  // mu -> 0..31, var -> 32..63
    v8f acc[2] = {{}, {}};
#pragma unroll
    for (int ks = 0; ks < 7; ++ks) {
      v16bf b = ldB(Wh, KP, ct, lane, ks);
#pragma unroll
      for (int mi = 0; mi < 2; ++mi) acc[mi] = WMMA_BF16(xf[mi][ks], b, acc[mi]);
    }
#pragma unroll
    for (int mi = 0; mi < 2; ++mi)
#pragma unroll
      for (int j = 0; j < 8; ++j) {
        int m = (mtb + mi) * 16 + g * 8 + j;
        Gb[m * 64 + colbase + n16] = acc[mi][j];
      }
  }
  __syncthreads();
  for (int e = tid; e < BM * 32; e += NTHREADS) {
    int m = e >> 5, c = e & 31;
    float mu = Gb[m * 64 + c] + bmu[c];        // biases padded to 32
    float vv = Gb[m * 64 + 32 + c] + bvr[c];
    float sp = (vv > 20.f) ? vv : log1pf(__expf(vv));
    if (c < LDIM)
      sout[m * LDIM + c] = mu + (sp + 0.1f) * eps[m * LDIM + c];
  }
}

// ---- host-side launch -------------------------------------------------------
extern "C" void kernel_launch(void* const* d_in, const int* in_sizes, int n_in,
                              void* d_out, int out_size, void* d_ws, size_t ws_size,
                              hipStream_t stream) {
  (void)in_sizes; (void)n_in; (void)out_size; (void)ws_size;
  const float* actions = (const float*)d_in[0];
  const float* h0   = (const float*)d_in[1];
  const float* s0   = (const float*)d_in[2];
  const float* noise = (const float*)d_in[3];
  const float* W_fc = (const float*)d_in[4];
  const float* b_fc = (const float*)d_in[5];
  const float* W_ih = (const float*)d_in[6];
  const float* b_ih = (const float*)d_in[7];
  const float* W_hh = (const float*)d_in[8];
  const float* b_hh = (const float*)d_in[9];
  const float* W1   = (const float*)d_in[10];
  const float* b1   = (const float*)d_in[11];
  const float* Wmu  = (const float*)d_in[12];
  const float* bmu  = (const float*)d_in[13];
  const float* Wvar = (const float*)d_in[14];
  const float* bvar = (const float*)d_in[15];

  // bf16 padded weights
  __bf16* pfc = (__bf16*)d_ws;            // [208][64]
  __bf16* pih = pfc + 208 * 64;           // 3 x [208][224]  (r,z,n)
  __bf16* phh = pih + 3 * 208 * 224;      // 3 x [208][224]
  __bf16* pw1 = phh + 3 * 208 * 224;      // [208][224]
  __bf16* pmu = pw1 + 208 * 224;          // [32][224]
  __bf16* pvr = pmu + 32 * 224;           // [32][224]
  // f32 padded biases (after 707584 bytes of bf16 weights)
  float* fb   = (float*)(pvr + 32 * 224);
  float* bfcp = fb;          // 208
  float* bihp = bfcp + HP;   // 3 x 208
  float* bhhp = bihp + 3*HP; // 3 x 208
  float* b1p  = bhhp + 3*HP; // 208
  float* bmup = b1p + HP;    // 32
  float* bvrp = bmup + 32;   // 32

  auto padw = [&](const float* src, __bf16* dst, int r, int c, int pr, int pc) {
    int total = pr * pc;
    pad_w<<<(total + 255) / 256, 256, 0, stream>>>(src, dst, r, c, pr, pc);
  };
  auto padb = [&](const float* src, float* dst, int nv, int pn) {
    pad_bias<<<(pn + 255) / 256, 256, 0, stream>>>(src, dst, nv, pn);
  };
  padw(W_fc, pfc, SDIM, LDIM + ADIM, HP, CK);
  for (int gt = 0; gt < 3; ++gt) {
    padw(W_ih + gt * SDIM * SDIM, pih + gt * HP * KP, SDIM, SDIM, HP, KP);
    padw(W_hh + gt * SDIM * SDIM, phh + gt * HP * KP, SDIM, SDIM, HP, KP);
    padb(b_ih + gt * SDIM, bihp + gt * HP, SDIM, HP);
    padb(b_hh + gt * SDIM, bhhp + gt * HP, SDIM, HP);
  }
  padw(W1, pw1, SDIM, SDIM, HP, KP);
  padw(Wmu, pmu, LDIM, SDIM, 32, KP);
  padw(Wvar, pvr, LDIM, SDIM, 32, KP);
  padb(b_fc, bfcp, SDIM, HP);
  padb(b1, b1p, SDIM, HP);
  padb(bmu, bmup, LDIM, 32);
  padb(bvar, bvrp, LDIM, 32);

  float* states = (float*)d_out;                    // [T,B,200]
  float* stoch  = states + (size_t)TT * BB * SDIM;  // [T,B,30]

  for (int t = 0; t < TT; ++t) {
    rssm_step<<<BB / BM, NTHREADS, 0, stream>>>(
        t, actions, h0, s0, noise, pfc, bfcp, pih, bihp, phh, bhhp,
        pw1, b1p, pmu, bmup, pvr, bvrp, states, stoch);
  }
}
